// GraphSAGE_23398981828718
// MI455X (gfx1250) — matrix operand order repaired and checked by reading, not compiled, not run on hardware
//
#include <hip/hip_runtime.h>
#include <math.h>

// Problem constants (match reference)
#define BB   16
#define NN   1024
#define KK   16
#define E1c  4
#define D0c  128
#define HHc  256
#define PPc  10
#define MMc  (BB * NN)                 // 16384 rows
#define EPSF 1.1920928955078125e-07f   // float32 eps

typedef __attribute__((ext_vector_type(16))) __bf16 v16bf;
typedef __attribute__((ext_vector_type(8)))  float  v8f;
typedef __attribute__((__vector_size__(16))) int    v4i_t;

#if defined(__AMDGCN__) && __has_builtin(__builtin_amdgcn_global_load_async_to_lds_b128)
#define HAVE_ASYNC_LDS 1
#else
#define HAVE_ASYNC_LDS 0
#endif

// ---------------------------------------------------------------------------
// 1) Embedding gather: state0[m, 0:128] = emb[node_feat[m], :]
// ---------------------------------------------------------------------------
__global__ void __launch_bounds__(128) k_embed(const int* __restrict__ feat,
                                               const float* __restrict__ emb,
                                               float* __restrict__ state0) {
  const int m = blockIdx.x;
  const int d = threadIdx.x;                // blockDim.x == 128 == D0
  const int f = feat[m];
  state0[(size_t)m * D0c + d] = emb[(size_t)f * D0c + d];
}

// ---------------------------------------------------------------------------
// 2) Repack weight W[Kd x 256] (f32 row-major) into WMMA B-operand tiles, bf16.
//    Tile (kt, nt): 32 lanes x 16 bf16 contiguous per lane.
//    h = lane/16 : element j<8  -> W[kt*32 + 8h + j][nt*16 + lane%16]
//                  element j>=8 -> W[kt*32 + 16 + 8h + (j-8)][nt*16 + lane%16]
// ---------------------------------------------------------------------------
__global__ void __launch_bounds__(256) k_pack_w(const float* __restrict__ W,
                                                unsigned short* __restrict__ dst_us,
                                                int Kd) {
  const int id    = blockIdx.x * blockDim.x + threadIdx.x;
  const int total = (Kd >> 5) * 16 * 512;   // KT * NT * (32 lanes * 16 elems)
  if (id >= total) return;
  const int j    = id & 15;
  const int lane = (id >> 4) & 31;
  const int tile = id >> 9;                 // kt*16 + nt
  const int nt   = tile & 15;
  const int kt   = tile >> 4;
  const int n    = nt * 16 + (lane & 15);
  const int k    = kt * 32 + ((j & 8) << 1) + ((lane >> 4) << 3) + (j & 7);
  __bf16* dst = reinterpret_cast<__bf16*>(dst_us);
  dst[id] = (__bf16)W[(size_t)k * HHc + n];
}

// ---------------------------------------------------------------------------
// 3) Neighbor mean-aggregation. One 256-thread block per node m = b*N + n.
//    msg[m, e*D + d] = bf16( mask[m] * mean_k state[b*N + nn[m,k,e], d] )
// ---------------------------------------------------------------------------
template <int D>
__global__ void __launch_bounds__(256) k_agg(const int* __restrict__ nn,
                                             const float* __restrict__ mask,
                                             const float* __restrict__ state,
                                             unsigned short* __restrict__ msg_us) {
  constexpr int V = D / 64;                 // floats per thread (2 or 4)
  const int m = blockIdx.x;
  const int b = m >> 10;                    // N = 1024
  __shared__ int sidx[KK * E1c];            // 64 indices for this node
  const int t = threadIdx.x;
  if (t < KK * E1c) sidx[t] = nn[(size_t)m * (KK * E1c) + t];
  __syncthreads();

  const int e     = t >> 6;                 // edge type 0..3
  const int dbase = (t & 63) * V;

  float acc[V];
#pragma unroll
  for (int v = 0; v < V; ++v) acc[v] = 0.0f;

#pragma unroll
  for (int k = 0; k < KK; ++k) {
    const int row = b * NN + sidx[k * E1c + e];
    const float* sp = state + (size_t)row * D + dbase;
    if (k + 1 < KK) {
      const int rown = b * NN + sidx[(k + 1) * E1c + e];
      __builtin_prefetch(state + (size_t)rown * D + dbase, 0, 1);
    }
    if constexpr (V == 4) {
      float4 x = *reinterpret_cast<const float4*>(sp);
      acc[0] += x.x; acc[1] += x.y; acc[2] += x.z; acc[3] += x.w;
    } else {
      float2 x = *reinterpret_cast<const float2*>(sp);
      acc[0] += x.x; acc[1] += x.y;
    }
  }

  const float mk = mask[m] * (1.0f / (float)KK);
  __bf16* op = reinterpret_cast<__bf16*>(msg_us) + (size_t)m * (E1c * D) + e * D + dbase;
#pragma unroll
  for (int v = 0; v < V; ++v) op[v] = (__bf16)(acc[v] * mk);
}

// ---------------------------------------------------------------------------
// 4) Fused bf16 WMMA GEMM + bias + ReLU + row-L2-normalize.
//    h[M,256] = relu(msg[M,Kd] @ Wpack + bias);  state = h / (||h||_2 + eps)
//    256 threads = 8 waves = 4 stripes of 16 rows; each stripe is computed by
//    a PAIR of waves, 8 N-tiles each (acc = 64 VGPRs/wave -> no spills).
//    B slices (16 KB per k-tile) double-buffer through LDS via the gfx1250
//    async global->LDS copy engine (ASYNCcnt): no staging VGPRs, the copy
//    hides entirely behind the 8-wmma burst.
// ---------------------------------------------------------------------------
__global__ void __launch_bounds__(256, 1) k_gemm_relu_norm(
    const unsigned short* __restrict__ msg_us,
    const unsigned short* __restrict__ wpack_us,
    const float* __restrict__ bias,
    float* __restrict__ stateOut,
    int Kd) {
  // 2 x 16KB bf16 B-slice buffers (slice = 16 tiles x 32 lanes x 16 bf16)
  __shared__ __align__(16) unsigned short sB[2 * 8192];
  __shared__ float sNorm[4][2][16];         // [stripe][wave-in-pair][row]

  const int t      = threadIdx.x;
  const int lane   = t & 31;
  const int wave   = t >> 5;                // 0..7
  const int stripe = wave >> 1;             // 0..3
  const int half   = wave & 1;              // which 8 N-tiles
  const int mt     = blockIdx.x * 4 + stripe;  // 16-row tile index
  const int KT     = Kd >> 5;               // K-tiles of 32
  const int hi     = lane >> 4;             // lane half (0/1)
  const int lo     = lane & 15;

  const __bf16* msg = reinterpret_cast<const __bf16*>(msg_us);

  // --- stage slice 0 into sB[buf 0]: 1024 uint4, 256 threads x 4 ---
  const uint4* wsrc = reinterpret_cast<const uint4*>(wpack_us);
  uint4*       sB4  = reinterpret_cast<uint4*>(sB);
#pragma unroll
  for (int i = 0; i < 4; ++i) sB4[i * 256 + t] = wsrc[i * 256 + t];
  __syncthreads();

  v8f acc[8] = {};                          // this wave's 8 N-tiles

  // A operand: row = mt*16 + lo; chunks at K offsets hi*8 and hi*8+16.
  const __bf16* arow = msg + (size_t)(mt * 16 + lo) * Kd + hi * 8;

  v16bf a_cur, a_nxt;
  {
    uint4* u = reinterpret_cast<uint4*>(&a_cur);
    u[0] = *reinterpret_cast<const uint4*>(arow);
    u[1] = *reinterpret_cast<const uint4*>(arow + 16);
  }

  for (int kt = 0; kt < KT; ++kt) {
    const int  cur  = kt & 1;
    const bool more = (kt + 1 < KT);
    if (more) {
#if HAVE_ASYNC_LDS
      // Async copy of next B slice directly into the other LDS buffer:
      // no VGPR staging; tracked by ASYNCcnt; hides behind the wmma burst.
      const uint4* src = wsrc + (size_t)(kt + 1) * 1024;
      uint4*       dst = sB4 + (cur ^ 1) * 1024;
#pragma unroll
      for (int i = 0; i < 4; ++i) {
        __builtin_amdgcn_global_load_async_to_lds_b128(
            (__attribute__((address_space(1))) v4i_t*)(src + i * 256 + t),
            (__attribute__((address_space(3))) v4i_t*)(dst + i * 256 + t),
            0, 0);
      }
#endif
      // Next A tile (register ping-pong; only 2 loads).
      const __bf16* pa = arow + (kt + 1) * 32;
      uint4* u = reinterpret_cast<uint4*>(&a_nxt);
      u[0] = *reinterpret_cast<const uint4*>(pa);
      u[1] = *reinterpret_cast<const uint4*>(pa + 16);
    }

    // This wave's 8 tiles start at N-tile half*8.
    const unsigned short* bbase = sB + cur * 8192 + (half * 8) * 512 + lane * 16;
#pragma unroll
    for (int j = 0; j < 8; ++j) {
      v16bf bv;
      uint4* u = reinterpret_cast<uint4*>(&bv);
      const uint4* p = reinterpret_cast<const uint4*>(bbase + j * 512);
      u[0] = p[0];
      u[1] = p[1];
      acc[j] = __builtin_amdgcn_wmma_f32_16x16x32_bf16(
          false, a_cur, false, bv, (short)0, acc[j], false, false);
    }

    if (more) {
#if HAVE_ASYNC_LDS
#if __has_builtin(__builtin_amdgcn_s_wait_asynccnt)
      __builtin_amdgcn_s_wait_asynccnt(0);
#else
      asm volatile("s_wait_asynccnt 0x0" ::: "memory");
#endif
#else
      // Fallback: direct copy after the burst (short-lived registers).
      const uint4* src = wsrc + (size_t)(kt + 1) * 1024;
      uint4*       dst = sB4 + (cur ^ 1) * 1024;
#pragma unroll
      for (int i = 0; i < 4; ++i) dst[i * 256 + t] = src[i * 256 + t];
#endif
      a_cur = a_nxt;
    }
    __syncthreads();   // all waves' copies visible; readers of `cur` done
  }

  // Epilogue: bias + relu, accumulate this wave's half-row sum of squares.
  float ss[8];
#pragma unroll
  for (int r = 0; r < 8; ++r) ss[r] = 0.0f;
#pragma unroll
  for (int j = 0; j < 8; ++j) {
    const int nt = half * 8 + j;
    const float bv = bias[nt * 16 + lo];
#pragma unroll
    for (int r = 0; r < 8; ++r) {
      float h = acc[j][r] + bv;
      h = h > 0.0f ? h : 0.0f;
      acc[j][r] = h;
      ss[r] += h * h;
    }
  }
  // Columns of row r (+8 for upper lane half) are spread across the 16 lanes
  // of each half: xor-reduce within the 16-lane group.
#pragma unroll
  for (int r = 0; r < 8; ++r) {
    float s = ss[r];
    s += __shfl_xor(s, 1, 32);
    s += __shfl_xor(s, 2, 32);
    s += __shfl_xor(s, 4, 32);
    s += __shfl_xor(s, 8, 32);
    ss[r] = s;                               // partial over this wave's 128 cols
  }
  // Combine the two waves' partials through LDS.
  if (lo == 0) {
#pragma unroll
    for (int r = 0; r < 8; ++r) sNorm[stripe][half][hi * 8 + r] = ss[r];
  }
  __syncthreads();
  float inv[8];
#pragma unroll
  for (int r = 0; r < 8; ++r) {
    const float s = sNorm[stripe][0][hi * 8 + r] + sNorm[stripe][1][hi * 8 + r];
    inv[r] = 1.0f / (sqrtf(s) + EPSF);
  }
  // Store normalized state (f32, [M, 256]).
#pragma unroll
  for (int j = 0; j < 8; ++j) {
    const int nt = half * 8 + j;
#pragma unroll
    for (int r = 0; r < 8; ++r) {
      const int row = mt * 16 + r + hi * 8;
      stateOut[(size_t)row * HHc + nt * 16 + lo] = acc[j][r] * inv[r];
    }
  }
}

// ---------------------------------------------------------------------------
// 5) Readout: score[b,p] = mean_n sigmoid(s·Watt + batt) * (s·Wout + bout)[p]
//    One block per batch; weights cached in LDS; LDS tree reduction.
// ---------------------------------------------------------------------------
__global__ void __launch_bounds__(256) k_out(const float* __restrict__ state,
                                             const float* __restrict__ Wout,
                                             const float* __restrict__ bout,
                                             const float* __restrict__ Watt,
                                             const float* __restrict__ batt,
                                             float* __restrict__ out) {
  const int b = blockIdx.x;
  const int t = threadIdx.x;
  __shared__ float sWout[HHc * PPc];        // 10 KB
  __shared__ float sWatt[HHc];
  __shared__ float red[256];
  for (int i = t; i < HHc * PPc; i += 256) sWout[i] = Wout[i];
  for (int i = t; i < HHc; i += 256) sWatt[i] = Watt[i];
  __syncthreads();

  float accp[PPc];
#pragma unroll
  for (int p = 0; p < PPc; ++p) accp[p] = 0.0f;

  const float ba = batt[0];
  for (int n = t; n < NN; n += 256) {
    const float* s = state + (size_t)(b * NN + n) * HHc;
    float y[PPc];
#pragma unroll
    for (int p = 0; p < PPc; ++p) y[p] = 0.0f;
    float a = 0.0f;
    for (int h = 0; h < HHc; ++h) {
      const float v = s[h];
      a += v * sWatt[h];
#pragma unroll
      for (int p = 0; p < PPc; ++p) y[p] += v * sWout[h * PPc + p];
    }
    const float att = 1.0f / (1.0f + __expf(-(a + ba)));
#pragma unroll
    for (int p = 0; p < PPc; ++p) accp[p] += att * (y[p] + bout[p]);
  }

#pragma unroll
  for (int p = 0; p < PPc; ++p) {
    red[t] = accp[p];
    __syncthreads();
    for (int s2 = 128; s2 > 0; s2 >>= 1) {
      if (t < s2) red[t] += red[t + s2];
      __syncthreads();
    }
    if (t == 0) out[b * PPc + p] = red[0] * (1.0f / (float)NN);
    __syncthreads();
  }
}

// ---------------------------------------------------------------------------
// Launch wrapper
// ---------------------------------------------------------------------------
extern "C" void kernel_launch(void* const* d_in, const int* in_sizes, int n_in,
                              void* d_out, int out_size, void* d_ws, size_t ws_size,
                              hipStream_t stream) {
  const int*   feat = (const int*)  d_in[0];   // [B,N]
  const int*   nn   = (const int*)  d_in[1];   // [B,N,K,E1]
  const float* mask = (const float*)d_in[2];   // [B,N,1]
  const float* emb  = (const float*)d_in[3];   // [NA,D0]
  const float* W0   = (const float*)d_in[4];   // [512,256]
  const float* b0   = (const float*)d_in[5];
  const float* W1   = (const float*)d_in[6];   // [1024,256]
  const float* b1   = (const float*)d_in[7];
  const float* Wout = (const float*)d_in[8];   // [256,10]
  const float* bout = (const float*)d_in[9];
  const float* Watt = (const float*)d_in[10];  // [256,1]
  const float* batt = (const float*)d_in[11];

  // Workspace layout (bytes):
  //   [0, 16M)       : state  f32 [M,256]  (layer0 input occupies [M,128] region)
  //   [16M, 48M)     : msg    bf16 [M,1024] max
  //   [48M, 48M+256K): W0 packed bf16
  //   [48M+1M, ...)  : W1 packed bf16
  char* ws = (char*)d_ws;
  float*          state = (float*)ws;
  unsigned short* msg   = (unsigned short*)(ws + ((size_t)16 << 20));
  unsigned short* w0p   = (unsigned short*)(ws + ((size_t)48 << 20));
  unsigned short* w1p   = (unsigned short*)(ws + ((size_t)48 << 20) + ((size_t)1 << 20));

  // 1) embed
  k_embed<<<MMc, 128, 0, stream>>>(feat, emb, state);
  // 2) pack weights to WMMA B layout (bf16)
  k_pack_w<<<(16 * 16 * 512 + 255) / 256, 256, 0, stream>>>(W0, w0p, 512);
  k_pack_w<<<(32 * 16 * 512 + 255) / 256, 256, 0, stream>>>(W1, w1p, 1024);
  // 3) layer 0
  k_agg<128><<<MMc, 256, 0, stream>>>(nn, mask, state, msg);
  k_gemm_relu_norm<<<MMc / 64, 256, 0, stream>>>(msg, w0p, b0, state, 512);
  // 4) layer 1
  k_agg<256><<<MMc, 256, 0, stream>>>(nn, mask, state, msg);
  k_gemm_relu_norm<<<MMc / 64, 256, 0, stream>>>(msg, w1p, b1, state, 1024);
  // 5) readout
  k_out<<<BB, 256, 0, stream>>>(state, Wout, bout, Watt, batt, (float*)d_out);
}
